// HeterogeneousGCN_12043088298066
// MI455X (gfx1250) — compile-verified, matrix operand order checked by reading, not run on hardware
//
#include <hip/hip_runtime.h>
#include <hip/hip_bf16.h>

typedef __attribute__((ext_vector_type(16))) __bf16 v16bf;
typedef __attribute__((ext_vector_type(8)))  __bf16 v8bf;
typedef __attribute__((ext_vector_type(8)))  float  v8f;

#define LDA 40          // bf16 per LDS tile row (32 data + 8 pad); 80B row = 16B aligned
#define TILE (64 * LDA) // one 64x32 bf16 tile (plus pad)

// 16x32 bf16 WMMA fragment load (A-layout; B tiles stored transposed in LDS so
// the same pattern applies). Lane (hf=lane>>4, lrow=lane&15) holds row `row`,
// K = {8*hf .. 8*hf+7} U {8*hf+16 .. 8*hf+23} -> two 16B LDS reads.
__device__ __forceinline__ v16bf frag_ld(const __bf16* t, int row, int hf) {
  const __bf16* p = t + row * LDA + hf * 8;
  v16bf f;
#pragma unroll
  for (int e = 0; e < 8; ++e) f[e] = p[e];
#pragma unroll
  for (int e = 0; e < 8; ++e) f[e + 8] = p[16 + e];
  return f;
}

__device__ __forceinline__ v8bf pk8(float4 f0, float4 f1) {
  v8bf o;
  o[0] = (__bf16)f0.x; o[1] = (__bf16)f0.y; o[2] = (__bf16)f0.z; o[3] = (__bf16)f0.w;
  o[4] = (__bf16)f1.x; o[5] = (__bf16)f1.y; o[6] = (__bf16)f1.z; o[7] = (__bf16)f1.w;
  return o;
}

// ---- staging helpers (128 threads) ---------------------------------------

// A tile 64x32 from f32 row-major: thread t -> row t>>1, 16 contiguous K.
__device__ __forceinline__ void stage_f32_A(__bf16* As_, const float* __restrict__ X,
                                            long rowBase, int k0, int t) {
  int r = t >> 1, kk0 = (t & 1) * 16;
  const float4* s = reinterpret_cast<const float4*>(X + (rowBase + r) * 256 + k0 + kk0);
  float4 f0 = s[0], f1 = s[1], f2 = s[2], f3 = s[3];
  v8bf* d = reinterpret_cast<v8bf*>(As_ + r * LDA + kk0);
  d[0] = pk8(f0, f1);
  d[1] = pk8(f2, f3);
}

// B tile [n][k] from f32 W[k][n]: float4 along n, 4-way scatter into LDS.
__device__ __forceinline__ void stage_f32_BT(__bf16* Bs_, const float* __restrict__ W,
                                             long kRowBase, int n0, int t) {
#pragma unroll
  for (int i = 0; i < 4; ++i) {
    int idx = i * 128 + t;
    int kk = idx >> 4, nb4 = (idx & 15) * 4;
    float4 f = *reinterpret_cast<const float4*>(W + (kRowBase + kk) * 256 + n0 + nb4);
    Bs_[(nb4 + 0) * LDA + kk] = (__bf16)f.x;
    Bs_[(nb4 + 1) * LDA + kk] = (__bf16)f.y;
    Bs_[(nb4 + 2) * LDA + kk] = (__bf16)f.z;
    Bs_[(nb4 + 3) * LDA + kk] = (__bf16)f.w;
  }
}

// A tile from non-transposed relation: A[r][kk] = w[kk] * rel[i0+r][k0+kk]
__device__ __forceinline__ void stage_rel_A(__bf16* As_, const int* __restrict__ rel,
                                            long relBase, long Nsrc, int i0,
                                            const float* __restrict__ wsrc, long wBase,
                                            int k0, int t) {
  int r = t >> 1, kk0 = (t & 1) * 16;
  const int4* rp = reinterpret_cast<const int4*>(rel + relBase + (long)(i0 + r) * Nsrc + k0 + kk0);
  const float4* wp = reinterpret_cast<const float4*>(wsrc + wBase + k0 + kk0);
  int4 a0 = rp[0], a1 = rp[1], a2 = rp[2], a3 = rp[3];
  float4 w0 = wp[0], w1 = wp[1], w2 = wp[2], w3 = wp[3];
  v8bf o0, o1;
  o0[0] = (__bf16)(w0.x * (float)a0.x); o0[1] = (__bf16)(w0.y * (float)a0.y);
  o0[2] = (__bf16)(w0.z * (float)a0.z); o0[3] = (__bf16)(w0.w * (float)a0.w);
  o0[4] = (__bf16)(w1.x * (float)a1.x); o0[5] = (__bf16)(w1.y * (float)a1.y);
  o0[6] = (__bf16)(w1.z * (float)a1.z); o0[7] = (__bf16)(w1.w * (float)a1.w);
  o1[0] = (__bf16)(w2.x * (float)a2.x); o1[1] = (__bf16)(w2.y * (float)a2.y);
  o1[2] = (__bf16)(w2.z * (float)a2.z); o1[3] = (__bf16)(w2.w * (float)a2.w);
  o1[4] = (__bf16)(w3.x * (float)a3.x); o1[5] = (__bf16)(w3.y * (float)a3.y);
  o1[6] = (__bf16)(w3.z * (float)a3.z); o1[7] = (__bf16)(w3.w * (float)a3.w);
  v8bf* d = reinterpret_cast<v8bf*>(As_ + r * LDA + kk0);
  d[0] = o0;
  d[1] = o1;
}

// A tile from transposed relation: A[r][kk] = w[kk] * rel[k0+kk][i0+r]
// (contiguous along r -> int4 loads, 4-way LDS scatter)
__device__ __forceinline__ void stage_relT_A(__bf16* As_, const int* __restrict__ rel,
                                             long relBase, long Ndst, int i0,
                                             const float* __restrict__ wsrc, long wBase,
                                             int k0, int t) {
  int kk = t >> 2, r0 = (t & 3) * 16;
  float wv = wsrc[wBase + k0 + kk];
  const int4* rp = reinterpret_cast<const int4*>(rel + relBase + (long)(k0 + kk) * Ndst + i0 + r0);
#pragma unroll
  for (int j = 0; j < 4; ++j) {
    int4 v = rp[j];
    As_[(r0 + j * 4 + 0) * LDA + kk] = (__bf16)(wv * (float)v.x);
    As_[(r0 + j * 4 + 1) * LDA + kk] = (__bf16)(wv * (float)v.y);
    As_[(r0 + j * 4 + 2) * LDA + kk] = (__bf16)(wv * (float)v.z);
    As_[(r0 + j * 4 + 3) * LDA + kk] = (__bf16)(wv * (float)v.w);
  }
}

// B tile [n][k] from bf16 info[k][n]: 16B loads along n, 8-way scatter.
__device__ __forceinline__ void stage_bf16_BT(__bf16* Bs_, const __bf16* __restrict__ info,
                                              long kRowBase, int n0, int t) {
#pragma unroll
  for (int i = 0; i < 2; ++i) {
    int idx = i * 128 + t;
    int kk = idx >> 3, nb8 = (idx & 7) * 8;
    v8bf v = *reinterpret_cast<const v8bf*>(info + (kRowBase + kk) * 256 + n0 + nb8);
#pragma unroll
    for (int j = 0; j < 8; ++j) Bs_[(nb8 + j) * LDA + kk] = v[j];
  }
}

// ---- small kernels --------------------------------------------------------

__global__ void zero_kernel(float* __restrict__ p, long n) {
  long i = (long)blockIdx.x * blockDim.x + threadIdx.x;
  if (i < n) p[i] = 0.f;
}

__global__ void nw_kernel(const float* __restrict__ X, const float* __restrict__ w,
                          const float* __restrict__ b, float* __restrict__ out, int total) {
  int node = blockIdx.x * (blockDim.x >> 5) + (threadIdx.x >> 5);
  int lane = threadIdx.x & 31;
  if (node >= total) return;
  const float* row = X + (long)node * 256;
  float s = 0.f;
  for (int d = lane; d < 256; d += 32) s += row[d] * w[d];
#pragma unroll
  for (int off = 16; off > 0; off >>= 1) s += __shfl_down(s, off, 32);
  if (lane == 0) out[node] = 1.f / (1.f + __expf(-(s + b[0])));
}

__global__ void rowsum_add_kernel(const int* __restrict__ rel, float* __restrict__ nb,
                                  int R, int C, int Bn) {
  int gw = blockIdx.x * (blockDim.x >> 5) + (threadIdx.x >> 5);
  int lane = threadIdx.x & 31;
  if (gw >= Bn * R) return;
  const int* p = rel + (long)gw * C;
  int s = 0;
  for (int j = lane; j < C; j += 32) s += p[j];
#pragma unroll
  for (int off = 16; off > 0; off >>= 1) s += __shfl_down(s, off, 32);
  if (lane == 0) nb[gw] += (float)s;
}

__global__ void colsum_add_kernel(const int* __restrict__ rel, float* __restrict__ nb,
                                  int R, int C, int Bn) {
  long idx = (long)blockIdx.x * blockDim.x + threadIdx.x;
  if (idx >= (long)Bn * C) return;
  int b = (int)(idx / C), c = (int)(idx % C);
  const int* p = rel + (long)b * R * C + c;
  int s = 0;
  for (int r = 0; r < R; ++r) s += p[(long)r * C];
  nb[idx] += (float)s;
}

// ---- WMMA GEMM kernels: 128 threads = 4 waves, block tile 64x64,
// wave (wr,wc) computes 32x32 via 4 accumulators. Double-buffered LDS. ------

__global__ __launch_bounds__(128) void lin_wmma_kernel(
    const float* __restrict__ X, const float* __restrict__ W,
    const float* __restrict__ bias, __bf16* __restrict__ Y, int M) {
  __shared__ __bf16 As[2 * TILE];
  __shared__ __bf16 Bs[2 * TILE];
  const int t = threadIdx.x;
  const long m0 = (long)blockIdx.x * 64;
  const int n0 = blockIdx.y * 64;
  const int wid = t >> 5, lane = t & 31;
  const int wr = wid & 1, wc = wid >> 1;
  const int hf = lane >> 4, lrow = lane & 15;

  v8f acc00 = {}, acc01 = {}, acc10 = {}, acc11 = {};
  stage_f32_A(As, X, m0, 0, t);
  stage_f32_BT(Bs, W, 0, n0, t);
  const int KT = 256 / 32;
  for (int kt = 0; kt < KT; ++kt) {
    __syncthreads();
    const __bf16* Ac = As + (kt & 1) * TILE;
    const __bf16* Bc = Bs + (kt & 1) * TILE;
    v16bf a0 = frag_ld(Ac, wr * 32 + lrow, hf);
    v16bf a1 = frag_ld(Ac, wr * 32 + 16 + lrow, hf);
    v16bf b0 = frag_ld(Bc, wc * 32 + lrow, hf);
    v16bf b1 = frag_ld(Bc, wc * 32 + 16 + lrow, hf);
    acc00 = __builtin_amdgcn_wmma_f32_16x16x32_bf16(false, a0, false, b0, (short)0, acc00, false, false);
    acc01 = __builtin_amdgcn_wmma_f32_16x16x32_bf16(false, a0, false, b1, (short)0, acc01, false, false);
    acc10 = __builtin_amdgcn_wmma_f32_16x16x32_bf16(false, a1, false, b0, (short)0, acc10, false, false);
    acc11 = __builtin_amdgcn_wmma_f32_16x16x32_bf16(false, a1, false, b1, (short)0, acc11, false, false);
    if (kt + 1 < KT) {
      __bf16* An = As + ((kt + 1) & 1) * TILE;
      __bf16* Bn = Bs + ((kt + 1) & 1) * TILE;
      stage_f32_A(An, X, m0, (kt + 1) * 32, t);
      stage_f32_BT(Bn, W, (long)(kt + 1) * 32, n0, t);
    }
  }
  const int n_a = n0 + wc * 32 + lrow, n_b = n_a + 16;
  const float ba = bias[n_a], bb = bias[n_b];
#pragma unroll
  for (int r = 0; r < 8; ++r) {
    long mA = m0 + wr * 32 + hf * 8 + r;   // D layout: M = r + 8*hf
    long mB = mA + 16;
    Y[mA * 256 + n_a] = (__bf16)(acc00[r] + ba);
    Y[mA * 256 + n_b] = (__bf16)(acc01[r] + bb);
    Y[mB * 256 + n_a] = (__bf16)(acc10[r] + ba);
    Y[mB * 256 + n_b] = (__bf16)(acc11[r] + bb);
  }
}

// out[b, rowOff+i, :] += sum_j (srcW[b,j] * rel(i,j)) * info[b,j,:]
__global__ __launch_bounds__(128) void agg_wmma_kernel(
    const int* __restrict__ rel, int relT,
    const float* __restrict__ srcW, const __bf16* __restrict__ info,
    float* __restrict__ out, int outRowOff, int outTotalRows,
    int Ndst, int Nsrc) {
  __shared__ __bf16 As[2 * TILE];
  __shared__ __bf16 Bs[2 * TILE];
  const int t = threadIdx.x;
  const int b  = blockIdx.z;
  const int i0 = blockIdx.x * 64;
  const int n0 = blockIdx.y * 64;
  const int wid = t >> 5, lane = t & 31;
  const int wr = wid & 1, wc = wid >> 1;
  const int hf = lane >> 4, lrow = lane & 15;
  const long relBase = (long)b * Ndst * Nsrc;
  const long wBase   = (long)b * Nsrc;
  const long iBase   = (long)b * Nsrc;   // info row base (x256 cols)

  v8f acc00 = {}, acc01 = {}, acc10 = {}, acc11 = {};
  if (!relT) stage_rel_A(As, rel, relBase, Nsrc, i0, srcW, wBase, 0, t);
  else       stage_relT_A(As, rel, relBase, Ndst, i0, srcW, wBase, 0, t);
  stage_bf16_BT(Bs, info, iBase, n0, t);
  const int KT = Nsrc / 32;
  for (int kt = 0; kt < KT; ++kt) {
    __syncthreads();
    const __bf16* Ac = As + (kt & 1) * TILE;
    const __bf16* Bc = Bs + (kt & 1) * TILE;
    v16bf a0 = frag_ld(Ac, wr * 32 + lrow, hf);
    v16bf a1 = frag_ld(Ac, wr * 32 + 16 + lrow, hf);
    v16bf b0 = frag_ld(Bc, wc * 32 + lrow, hf);
    v16bf b1 = frag_ld(Bc, wc * 32 + 16 + lrow, hf);
    acc00 = __builtin_amdgcn_wmma_f32_16x16x32_bf16(false, a0, false, b0, (short)0, acc00, false, false);
    acc01 = __builtin_amdgcn_wmma_f32_16x16x32_bf16(false, a0, false, b1, (short)0, acc01, false, false);
    acc10 = __builtin_amdgcn_wmma_f32_16x16x32_bf16(false, a1, false, b0, (short)0, acc10, false, false);
    acc11 = __builtin_amdgcn_wmma_f32_16x16x32_bf16(false, a1, false, b1, (short)0, acc11, false, false);
    if (kt + 1 < KT) {
      __bf16* An = As + ((kt + 1) & 1) * TILE;
      __bf16* Bn = Bs + ((kt + 1) & 1) * TILE;
      int k0 = (kt + 1) * 32;
      if (!relT) stage_rel_A(An, rel, relBase, Nsrc, i0, srcW, wBase, k0, t);
      else       stage_relT_A(An, rel, relBase, Ndst, i0, srcW, wBase, k0, t);
      stage_bf16_BT(Bn, info, iBase + k0, n0, t);
    }
  }
  const int n_a = n0 + wc * 32 + lrow, n_b = n_a + 16;
#pragma unroll
  for (int r = 0; r < 8; ++r) {
    long rowA = (long)b * outTotalRows + outRowOff + i0 + wr * 32 + hf * 8 + r;
    long rowB = rowA + 16;
    out[rowA * 256 + n_a] += acc00[r];
    out[rowA * 256 + n_b] += acc01[r];
    out[rowB * 256 + n_a] += acc10[r];
    out[rowB * 256 + n_b] += acc11[r];
  }
}

// out = relu(self + out / max(nb,1)), in place on the destination slice
__global__ void combine_kernel(float* __restrict__ out, const __bf16* __restrict__ selfLin,
                               const float* __restrict__ nb, int Ndst, int rowOff,
                               int totalRows, int Bn) {
  long idx = (long)blockIdx.x * blockDim.x + threadIdx.x;
  long total = (long)Bn * Ndst * 256;
  if (idx >= total) return;
  int d = (int)(idx & 255);
  long ni = idx >> 8;
  int b = (int)(ni / Ndst), i = (int)(ni % Ndst);
  float nbv = nb[ni];
  nbv = nbv >= 1.f ? nbv : 1.f;
  long o = ((long)b * totalRows + rowOff + i) * 256 + d;
  float v = (float)selfLin[idx] + out[o] / nbv;
  out[o] = v > 0.f ? v : 0.f;
}

extern "C" void kernel_launch(void* const* d_in, const int* in_sizes, int n_in,
                              void* d_out, int out_size, void* d_ws, size_t ws_size,
                              hipStream_t stream) {
  (void)in_sizes; (void)n_in; (void)ws_size;
  const int B = 8, S = 1024, E = 2048, V = 1024, Q = 64, QE = 128, QV = 64;
  const int TOT = S + E + V + Q + QE + QV;            // 4352

  const float* sent = (const float*)d_in[0];
  const float* ent  = (const float*)d_in[1];
  const float* val  = (const float*)d_in[2];
  const int*   svR  = (const int*)d_in[3];
  const int*   eeR  = (const int*)d_in[4];
  const int*   evR  = (const int*)d_in[5];
  const int*   seR  = (const int*)d_in[6];
  const float* qn   = (const float*)d_in[10];
  const float* qen  = (const float*)d_in[11];
  const float* qvn  = (const float*)d_in[12];
  const int*   qeR  = (const int*)d_in[13];
  const int*   qvR  = (const int*)d_in[14];
  const int*   qeeR = (const int*)d_in[15];
  const int*   qevR = (const int*)d_in[16];
  const float* nwW  = (const float*)d_in[19];
  const float* nwB  = (const float*)d_in[20];
  auto Wfc = [&](int k) { return (const float*)d_in[21 + 2 * k]; };
  auto Bfc = [&](int k) { return (const float*)d_in[22 + 2 * k]; };
  float* out = (float*)d_out;

  char* wsp = (char*)d_ws; size_t off = 0;
  auto alloc = [&](size_t bytes) -> void* {
    void* p = wsp + off; off += (bytes + 255) & ~(size_t)255; return p;
  };
  float* ew  = (float*)alloc((size_t)B * E  * 4);
  float* sw  = (float*)alloc((size_t)B * S  * 4);
  float* vw  = (float*)alloc((size_t)B * V  * 4);
  float* qw  = (float*)alloc((size_t)B * Q  * 4);
  float* qew = (float*)alloc((size_t)B * QE * 4);
  float* qvw = (float*)alloc((size_t)B * QV * 4);
  float* nbSlab = (float*)alloc((size_t)B * TOT * 4);
  float* s_nb  = nbSlab;
  float* e_nb  = s_nb + (size_t)B * S;
  float* v_nb  = e_nb + (size_t)B * E;
  float* q_nb  = v_nb + (size_t)B * V;
  float* qe_nb = q_nb + (size_t)B * Q;
  float* qv_nb = qe_nb + (size_t)B * QE;

  const int fcN[20]    = {E, V, S, QE, QV, Q, E, V, V, E, S, S, E, QE, QV, QV, QE, Q, Q, QE};
  const float* fcX[20] = {ent, val, sent, qen, qvn, qn, ent, val, val, ent, sent,
                          sent, ent, qen, qvn, qvn, qen, qn, qn, qen};
  __bf16* lin[20];
  for (int k = 0; k < 20; ++k) lin[k] = (__bf16*)alloc((size_t)B * fcN[k] * 256 * 2);

  { long n = (long)out_size; zero_kernel<<<(n + 255) / 256, 256, 0, stream>>>(out, n); }
  { long n = (long)B * TOT;  zero_kernel<<<(n + 255) / 256, 256, 0, stream>>>(nbSlab, n); }

  auto launch_nw = [&](const float* X, float* o, int total) {
    int blocks = (total * 32 + 255) / 256;
    nw_kernel<<<blocks, 256, 0, stream>>>(X, nwW, nwB, o, total);
  };
  launch_nw(ent, ew, B * E);   launch_nw(sent, sw, B * S); launch_nw(val, vw, B * V);
  launch_nw(qen, qew, B * QE); launch_nw(qn, qw, B * Q);   launch_nw(qvn, qvw, B * QV);

  auto rowsum = [&](const int* r, float* nbp, int R, int C) {
    int blocks = (B * R * 32 + 255) / 256;
    rowsum_add_kernel<<<blocks, 256, 0, stream>>>(r, nbp, R, C, B);
  };
  auto colsum = [&](const int* r, float* nbp, int R, int C) {
    long total = (long)B * C;
    colsum_add_kernel<<<(total + 255) / 256, 256, 0, stream>>>(r, nbp, R, C, B);
  };
  rowsum(seR, s_nb, S, E);  rowsum(svR, s_nb, S, V);
  colsum(seR, e_nb, S, E);  rowsum(eeR, e_nb, E, E);  rowsum(evR, e_nb, E, V);
  colsum(svR, v_nb, S, V);  colsum(evR, v_nb, E, V);
  rowsum(qeR, q_nb, Q, QE); rowsum(qvR, q_nb, Q, QV);
  colsum(qeR, qe_nb, Q, QE); rowsum(qeeR, qe_nb, QE, QE); rowsum(qevR, qe_nb, QE, QV);
  colsum(qvR, qv_nb, Q, QV); colsum(qevR, qv_nb, QE, QV);

  for (int k = 0; k < 20; ++k) {
    int M = B * fcN[k];
    dim3 g(M / 64, 4, 1);
    lin_wmma_kernel<<<g, 128, 0, stream>>>(fcX[k], Wfc(k), Bfc(k), lin[k], M);
  }

  const int offS = 0, offE = S, offV = S + E, offQ = S + E + V,
            offQE = S + E + V + Q, offQV = S + E + V + Q + QE;
  auto agg = [&](const int* relp, int relT, const float* wsrc, const __bf16* info,
                 int rowOff, int Ndst, int Nsrc) {
    dim3 g(Ndst / 64, 4, B);
    agg_wmma_kernel<<<g, 128, 0, stream>>>(relp, relT, wsrc, info, out, rowOff, TOT, Ndst, Nsrc);
  };
  agg(seR, 0, ew,  lin[6],  offS,  S,  E);   // sentence <- entity
  agg(svR, 0, vw,  lin[7],  offS,  S,  V);   // sentence <- value
  agg(seR, 1, sw,  lin[10], offE,  E,  S);   // entity   <- sentence (se^T)
  agg(eeR, 0, ew,  lin[9],  offE,  E,  E);   // entity   <- entity
  agg(evR, 0, vw,  lin[8],  offE,  E,  V);   // entity   <- value
  agg(svR, 1, sw,  lin[11], offV,  V,  S);   // value    <- sentence (sv^T)
  agg(evR, 1, ew,  lin[12], offV,  V,  E);   // value    <- entity   (ev^T)
  agg(qeR, 0, qew, lin[13], offQ,  Q,  QE);  // question <- q_entity
  agg(qvR, 0, qvw, lin[14], offQ,  Q,  QV);  // question <- q_value
  agg(qeR, 1, qw,  lin[17], offQE, QE, Q);   // q_entity <- question (qe^T)
  agg(qeeR,0, qew, lin[16], offQE, QE, QE);  // q_entity <- q_entity
  agg(qevR,0, qvw, lin[15], offQE, QE, QV);  // q_entity <- q_value
  agg(qvR, 1, qw,  lin[18], offQV, QV, Q);   // q_value  <- question (qv^T)
  agg(qevR,1, qew, lin[19], offQV, QV, QE);  // q_value  <- q_entity (qev^T)

  auto comb = [&](const __bf16* selfb, const float* nbp, int rowOff, int Ndst) {
    long total = (long)B * Ndst * 256;
    combine_kernel<<<(total + 255) / 256, 256, 0, stream>>>(out, selfb, nbp, Ndst, rowOff, TOT, B);
  };
  comb(lin[2], s_nb,  offS,  S);
  comb(lin[0], e_nb,  offE,  E);
  comb(lin[1], v_nb,  offV,  V);
  comb(lin[5], q_nb,  offQ,  Q);
  comb(lin[3], qe_nb, offQE, QE);
  comb(lin[4], qv_nb, offQV, QV);
}
